// MemNet_18391049961827
// MI455X (gfx1250) — matrix operand order repaired and checked by reading, not compiled, not run on hardware
//
#include <hip/hip_runtime.h>
#include <stdint.h>

// ---- problem dims ----
#define B_   32
#define T_   128
#define V_   10000
#define E_   256
#define H_   512
#define S_   128
#define D_   128
#define NH_  4
#define KTOP_ 8
#define G3H_ (3 * H_)   // 1536
#define ED_  (E_ + D_)  // 384
#define EPS_ 1e-8f

typedef __bf16 bf16;
typedef __attribute__((ext_vector_type(16))) __bf16 v16bf;
typedef __attribute__((ext_vector_type(8)))  float  v8f;

union Frag {
    v16bf    v;
    uint32_t w[8];
    bf16     e[16];
};

// ---- workspace layout (bytes, all 256-aligned) ----
static constexpr size_t OFF_WIH   = 0;                         // 1536*384  bf16 (packed)
static constexpr size_t OFF_WHH   = 1179648;                   // 1536*512  bf16 (packed)
static constexpr size_t OFF_WOUT  = 2752512;                   // 512*10000 bf16 (packed)
static constexpr size_t OFF_STATE = 12992512;
static constexpr size_t OFF_H     = OFF_STATE;                 // 32*512  f32
static constexpr size_t OFF_RV    = OFF_H  + 65536;            // 32*128  f32
static constexpr size_t OFF_M     = OFF_RV + 16384;            // 32*128*128 f32
static constexpr size_t OFF_GI    = OFF_M  + 2097152;          // 32*1536 f32
static constexpr size_t OFF_GH    = OFF_GI + 196608;           // 32*1536 f32
static constexpr size_t OFF_HBF   = OFF_GH + 196608;           // 32*512  bf16
static constexpr size_t OFF_INPBF = OFF_HBF + 32768;           // 32*384  bf16
static constexpr size_t OFF_END   = OFF_INPBF + 24576;
static constexpr size_t STATE_WORDS = (OFF_END - OFF_STATE) / 4;

__device__ __forceinline__ float sigmoidf_(float x) { return 1.f / (1.f + expf(-x)); }

// ---------------------------------------------------------------------------
// Fused f32->bf16 convert + swizzle into this file's WMMA B-fragment order (k contiguous per lane half).
// Packed index: ((n_tile*KSTEPS + kk)*32 + lane)*16 + j  maps to B(k,n) with
//   n = n_tile*16 + (lane&15),  k = kk*32 + 16*(lane>>4) + j
// so the GEMM's per-lane B fetch is 32 contiguous bytes (2x b128 per lane,
// 1 KB contiguous per wave per k-step).
//   transposed=1: source W stored [N,Kdim] row-major (we multiply by W^T)
//   transposed=0: source W stored [Kdim,N] row-major
// ---------------------------------------------------------------------------
__global__ void k_pack_b(const float* __restrict__ W, bf16* __restrict__ P,
                         int N, int Kdim, int transposed)
{
    const int ksteps = Kdim >> 5;
    const int total  = N * Kdim;
    for (int i = blockIdx.x * blockDim.x + threadIdx.x; i < total;
         i += gridDim.x * blockDim.x) {
        const int j    = i & 15;
        const int lane = (i >> 4) & 31;
        const int blk  = i >> 9;             // n_tile*ksteps + kk
        const int kk   = blk % ksteps;
        const int n_t  = blk / ksteps;
        const int n = (n_t << 4) + (lane & 15);
        const int k = (kk << 5) + ((lane >> 4) << 4) + j;
        const float v = transposed ? W[(size_t)n * Kdim + k]
                                   : W[(size_t)k * N + n];
        P[i] = (bf16)v;
    }
}

__global__ void k_zero_u32(uint32_t* __restrict__ p, int n) {
    for (int i = blockIdx.x * blockDim.x + threadIdx.x; i < n; i += gridDim.x * blockDim.x)
        p[i] = 0u;
}

// ---------------------------------------------------------------------------
// A fragment (16x32 bf16): lanes 0-15 hold K {0..15}, lanes 16-31 hold
// K {16..31} of rows M=lane&15.  Two b128 loads per lane.
// ---------------------------------------------------------------------------
__device__ __forceinline__ v16bf load_a_frag(const bf16* __restrict__ pA, int kk)
{
    Frag fa;
    const uint32_t* a0 = (const uint32_t*)(pA + (kk << 5));
    const uint32_t* a1 = (const uint32_t*)(pA + (kk << 5) + 8);
#pragma unroll
    for (int i = 0; i < 4; ++i) { fa.w[i] = a0[i]; fa.w[4 + i] = a1[i]; }
    return fa.v;
}

__device__ __forceinline__ v16bf load_b_frag(const bf16* __restrict__ pB, int kk)
{
    Frag fb;
    const uint32_t* b = (const uint32_t*)(pB + (size_t)kk * 512);
#pragma unroll
    for (int i = 0; i < 8; ++i) fb.w[i] = b[i];
    return fb.v;
}

// ---------------------------------------------------------------------------
// bf16 WMMA GEMM: C[32,N] = A[32,Kdim] * B(packed) + bias
// One wave computes a 16x32 slab (1 M-tile x 2 N-tiles) with A-fragment reuse;
// K loop fully unrolled (KSTEPS template) -> 2*KSTEPS static v_wmma.
// Wave index is forced into an SGPR (readfirstlane) so every guard is a
// scalar branch and EXEC stays all-ones around WMMA (ISA 7.12 restriction).
// ---------------------------------------------------------------------------
template <int KSTEPS>
__global__ void k_wmma_gemm(const bf16* __restrict__ A, int lda,
                            const bf16* __restrict__ P,
                            const float* __restrict__ bias,
                            float* __restrict__ C, long long c_stride, int N)
{
    const int lane = threadIdx.x & 31;
    const int wave = __builtin_amdgcn_readfirstlane(
        (int)(blockIdx.x * (blockDim.x >> 5) + (threadIdx.x >> 5)));
    const int ntiles = N >> 4;
    const int npairs = (ntiles + 1) >> 1;
    if (wave >= 2 * npairs) return;          // scalar branch; M rows = 32 -> 2 row tiles
    const int mtile  = wave / npairs;
    const int ntile0 = (wave % npairs) << 1;
    const bool has2  = (ntile0 + 1) < ntiles;   // scalar condition

    const int mrow = (mtile << 4) + (lane & 15);
    const int hi   = (lane >> 4) & 1;

    const bf16* pA  = A + (size_t)mrow * lda + (hi ? 16 : 0);
    const bf16* pB0 = P + (size_t)ntile0 * KSTEPS * 512 + lane * 16;
    const bf16* pB1 = pB0 + (size_t)KSTEPS * 512;

    v8f acc0 = {0.f, 0.f, 0.f, 0.f, 0.f, 0.f, 0.f, 0.f};
    v8f acc1 = {0.f, 0.f, 0.f, 0.f, 0.f, 0.f, 0.f, 0.f};

    if (has2) {
#pragma unroll
        for (int kk = 0; kk < KSTEPS; ++kk) {
            v16bf fa  = load_a_frag(pA, kk);
            v16bf fb0 = load_b_frag(pB0, kk);
            v16bf fb1 = load_b_frag(pB1, kk);
            acc0 = __builtin_amdgcn_wmma_f32_16x16x32_bf16(false, fa, false, fb0,
                                                           (short)0, acc0, false, false);
            acc1 = __builtin_amdgcn_wmma_f32_16x16x32_bf16(false, fa, false, fb1,
                                                           (short)0, acc1, false, false);
        }
    } else {
#pragma unroll
        for (int kk = 0; kk < KSTEPS; ++kk) {
            v16bf fa  = load_a_frag(pA, kk);
            v16bf fb0 = load_b_frag(pB0, kk);
            acc0 = __builtin_amdgcn_wmma_f32_16x16x32_bf16(false, fa, false, fb0,
                                                           (short)0, acc0, false, false);
        }
    }

    // C rows: lanes 0-15 -> M 0-7, lanes 16-31 -> M 8-15 of the tile
    const int   mbase = (mtile << 4) + (hi ? 8 : 0);
    const int   n0    = (ntile0 << 4) + (lane & 15);
    const float bv0   = bias ? bias[n0] : 0.f;
#pragma unroll
    for (int r = 0; r < 8; ++r)
        C[(size_t)(mbase + r) * (size_t)c_stride + n0] = acc0[r] + bv0;
    if (has2) {
        const int   n1  = n0 + 16;
        const float bv1 = bias ? bias[n1] : 0.f;
#pragma unroll
        for (int r = 0; r < 8; ++r)
            C[(size_t)(mbase + r) * (size_t)c_stride + n1] = acc1[r] + bv1;
    }
}

// ---------------------------------------------------------------------------
// Build controller input: inp = [embed[x_t] | rv]  (bf16)
// ---------------------------------------------------------------------------
__global__ void k_build_inp(const int* __restrict__ seq, const float* __restrict__ embed,
                            const float* __restrict__ rv, bf16* __restrict__ inp_bf, int t)
{
    int idx = blockIdx.x * blockDim.x + threadIdx.x;
    if (idx >= B_ * ED_) return;
    int b = idx / ED_, j = idx % ED_;
    float v = (j < E_) ? embed[(size_t)seq[b * T_ + t] * E_ + j]
                       : rv[(size_t)b * D_ + (j - E_)];
    inp_bf[idx] = (bf16)v;
}

// ---------------------------------------------------------------------------
// GRU gate combine: h_new = (1-z)*n + z*h   (PyTorch gate order r,z,n)
// ---------------------------------------------------------------------------
__global__ void k_hnew(const float* __restrict__ gi, const float* __restrict__ gh,
                       float* __restrict__ h, bf16* __restrict__ h_bf)
{
    int idx = blockIdx.x * blockDim.x + threadIdx.x;
    if (idx >= B_ * H_) return;
    int b = idx / H_, j = idx % H_;
    const float* gib = gi + (size_t)b * G3H_;
    const float* ghb = gh + (size_t)b * G3H_;
    float r  = sigmoidf_(gib[j] + ghb[j]);
    float z  = sigmoidf_(gib[H_ + j] + ghb[H_ + j]);
    float n  = tanhf(gib[2 * H_ + j] + r * ghb[2 * H_ + j]);
    float hn = (1.f - z) * n + z * h[idx];
    h[idx]   = hn;
    h_bf[idx] = (bf16)hn;
}

// ---------------------------------------------------------------------------
// top-k (order statistic) sharpened softmax over S=128 slots, one slot/thread.
// rank via total order (value desc, index asc) reproduces lax.top_k's kth value.
// ---------------------------------------------------------------------------
__device__ float topk_softmax_w(float sc, int tid, float* s_buf, float* s_scal)
{
    s_buf[tid] = sc;
    __syncthreads();
    int rank = 0;
    for (int j = 0; j < S_; ++j) {
        float o = s_buf[j];
        rank += (o > sc) || (o == sc && j < tid);
    }
    if (rank == KTOP_ - 1) s_scal[0] = sc;   // kth largest
    if (rank == 0)         s_scal[1] = sc;   // max
    __syncthreads();
    const float kth = s_scal[0];
    const float mx  = s_scal[1];
    float e = (sc >= kth) ? expf(sc - mx) : 0.f;
    s_buf[tid] = e;
    __syncthreads();
    if (tid == 0) {
        float sm = 0.f;
        for (int j = 0; j < S_; ++j) sm += s_buf[j];
        s_scal[2] = sm;
    }
    __syncthreads();
    float w = e / s_scal[2];
    __syncthreads();
    return w;
}

// ---------------------------------------------------------------------------
// Memory read (NH heads, cosine + topk softmax) and erase/add write.
// One workgroup (128 threads = 4 waves) per batch row.
// ---------------------------------------------------------------------------
__global__ void k_memory(const float* __restrict__ h,
                         float* __restrict__ M,
                         float* __restrict__ rv,
                         const float* __restrict__ W_rk,
                         const float* __restrict__ W_wk,
                         const float* __restrict__ W_wv,
                         const float* __restrict__ W_er,
                         const float* __restrict__ W_ag,
                         const float* __restrict__ beta_read,
                         const float* __restrict__ beta_write)
{
    __shared__ float s_invn[S_];
    __shared__ float s_rkn[NH_ * D_];
    __shared__ float s_wkn[D_];
    __shared__ float s_wv[D_];
    __shared__ float s_er[D_];
    __shared__ float s_buf[S_];
    __shared__ float s_w[S_];
    __shared__ float s_scal[12];
    __shared__ float s_ag;

    const int b   = blockIdx.x;
    const int tid = threadIdx.x;               // 0..127
    float* Mb = M + (size_t)b * S_ * D_;
    const float* hb = h + (size_t)b * H_;

    float br = beta_read[0];  br = fminf(fmaxf(log1pf(expf(br)), 1.f), 20.f);
    float bw = beta_write[0]; bw = fminf(fmaxf(log1pf(expf(bw)), 1.f), 20.f);

    // inverse row norms of (old) M
    {
        float ss = 0.f;
        const float* row = Mb + (size_t)tid * D_;
        for (int d = 0; d < D_; ++d) ss += row[d] * row[d];
        s_invn[tid] = 1.f / (sqrtf(ss) + EPS_);
    }
    // read keys rk: 4 outputs per thread
    for (int q = 0; q < 4; ++q) {
        int j = tid * 4 + q;
        float a = 0.f;
        for (int k = 0; k < H_; ++k) a += hb[k] * W_rk[(size_t)k * (NH_ * D_) + j];
        s_rkn[j] = a;
    }
    // write key / value / erase
    {
        float a0 = 0.f, a1 = 0.f, a2 = 0.f;
        for (int k = 0; k < H_; ++k) {
            float hv = hb[k];
            a0 += hv * W_wk[(size_t)k * D_ + tid];
            a1 += hv * W_wv[(size_t)k * D_ + tid];
            a2 += hv * W_er[(size_t)k * D_ + tid];
        }
        s_wkn[tid] = a0; s_wv[tid] = a1; s_er[tid] = sigmoidf_(a2);
    }
    if (tid == 0) {
        float a = 0.f;
        for (int k = 0; k < H_; ++k) a += hb[k] * W_ag[k];
        s_ag = sigmoidf_(a);
    }
    __syncthreads();

    // key inverse norms: heads 0..3 in s_scal[4..7], write key in s_scal[8]
    if (tid < NH_) {
        float ss = 0.f;
        for (int d = 0; d < D_; ++d) { float v = s_rkn[tid * D_ + d]; ss += v * v; }
        s_scal[4 + tid] = 1.f / (sqrtf(ss) + EPS_);
    }
    if (tid == NH_) {
        float ss = 0.f;
        for (int d = 0; d < D_; ++d) { float v = s_wkn[d]; ss += v * v; }
        s_scal[8] = 1.f / (sqrtf(ss) + EPS_);
    }
    __syncthreads();
    for (int q = 0; q < 4; ++q) {
        int j = tid * 4 + q;
        s_rkn[j] *= s_scal[4 + (j >> 7)];
    }
    s_wkn[tid] *= s_scal[8];
    __syncthreads();

    // ---- multi-head content read (uses old M) ----
    float rvacc = 0.f;
    for (int hh = 0; hh < NH_; ++hh) {
        const float* key = s_rkn + hh * D_;
        const float* row = Mb + (size_t)tid * D_;
        float sc = 0.f;
        for (int d = 0; d < D_; ++d) sc += key[d] * row[d];
        sc *= br * s_invn[tid];
        float w = topk_softmax_w(sc, tid, s_buf, s_scal);
        s_w[tid] = w;
        __syncthreads();
        float a = 0.f;
        for (int s = 0; s < S_; ++s) a += s_w[s] * Mb[(size_t)s * D_ + tid];
        rvacc += a;
        __syncthreads();
    }
    rv[(size_t)b * D_ + tid] = rvacc * (1.f / NH_);

    // ---- erase/add write ----
    {
        const float* row = Mb + (size_t)tid * D_;
        float sc = 0.f;
        for (int d = 0; d < D_; ++d) sc += s_wkn[d] * row[d];
        sc *= bw * s_invn[tid];
        float w = topk_softmax_w(sc, tid, s_buf, s_scal);
        s_w[tid] = w;
        __syncthreads();
        float erd = s_er[tid], wvd = s_wv[tid], ag = s_ag;
        for (int s = 0; s < S_; ++s) {
            size_t i = (size_t)s * D_ + tid;
            float m = Mb[i];
            Mb[i] = m * (1.f - s_w[s] * erd) + ag * s_w[s] * wvd;
        }
    }
}

// ---------------------------------------------------------------------------
extern "C" void kernel_launch(void* const* d_in, const int* in_sizes, int n_in,
                              void* d_out, int out_size, void* d_ws, size_t ws_size,
                              hipStream_t stream)
{
    const int*   seq    = (const int*)  d_in[0];
    const float* embed  = (const float*)d_in[1];
    const float* W_ih   = (const float*)d_in[2];
    const float* b_ih   = (const float*)d_in[3];
    const float* W_hh   = (const float*)d_in[4];
    const float* b_hh   = (const float*)d_in[5];
    const float* W_out  = (const float*)d_in[6];
    const float* b_out  = (const float*)d_in[7];
    const float* W_rk   = (const float*)d_in[8];
    const float* W_wk   = (const float*)d_in[9];
    const float* W_wv   = (const float*)d_in[10];
    const float* W_er   = (const float*)d_in[11];
    const float* W_ag   = (const float*)d_in[12];
    const float* beta_r = (const float*)d_in[13];
    const float* beta_w = (const float*)d_in[14];
    float* out = (float*)d_out;
    char*  ws  = (char*)d_ws;

    bf16*  Wih_p   = (bf16*)(ws + OFF_WIH);
    bf16*  Whh_p   = (bf16*)(ws + OFF_WHH);
    bf16*  Wout_p  = (bf16*)(ws + OFF_WOUT);
    float* hp      = (float*)(ws + OFF_H);
    float* rvp     = (float*)(ws + OFF_RV);
    float* Mp      = (float*)(ws + OFF_M);
    float* gip     = (float*)(ws + OFF_GI);
    float* ghp     = (float*)(ws + OFF_GH);
    bf16*  h_bf    = (bf16*)(ws + OFF_HBF);
    bf16*  inp_bf  = (bf16*)(ws + OFF_INPBF);

    // one-time per launch: pack weights into WMMA fragment layout, zero state
    k_pack_b<<<512,  256, 0, stream>>>(W_ih,  Wih_p,  G3H_, ED_, 1);
    k_pack_b<<<512,  256, 0, stream>>>(W_hh,  Whh_p,  G3H_, H_,  1);
    k_pack_b<<<2048, 256, 0, stream>>>(W_out, Wout_p, V_,   H_,  0);
    k_zero_u32<<<1024, 256, 0, stream>>>((uint32_t*)(ws + OFF_STATE), (int)STATE_WORDS);

    // waves = 2 * ceil(ntiles/2); 8 waves per 256-thread block
    const int gates_waves  = 2 * ((G3H_ / 16 + 1) / 2);   // 96
    const int logits_waves = 2 * ((V_   / 16 + 1) / 2);   // 626
    const int gates_blocks  = (gates_waves  + 7) / 8;     // 12
    const int logits_blocks = (logits_waves + 7) / 8;     // 79

    for (int t = 0; t < T_; ++t) {
        k_build_inp<<<(B_ * ED_ + 255) / 256, 256, 0, stream>>>(seq, embed, rvp, inp_bf, t);
        // gi = inp @ W_ih^T + b_ih       (K = 384 -> 12 k-steps)
        k_wmma_gemm<12><<<gates_blocks, 256, 0, stream>>>(inp_bf, ED_, Wih_p, b_ih,
                                                          gip, (long long)G3H_, G3H_);
        // gh = h @ W_hh^T + b_hh         (K = 512 -> 16 k-steps)
        k_wmma_gemm<16><<<gates_blocks, 256, 0, stream>>>(h_bf, H_, Whh_p, b_hh,
                                                          ghp, (long long)G3H_, G3H_);
        k_hnew<<<(B_ * H_ + 255) / 256, 256, 0, stream>>>(gip, ghp, hp, h_bf);
        // logits = h_new @ W_out + b_out -> out[:, t, :]
        k_wmma_gemm<16><<<logits_blocks, 256, 0, stream>>>(h_bf, H_, Wout_p, b_out,
                                                           out + (size_t)t * V_,
                                                           (long long)T_ * V_, V_);
        k_memory<<<B_, S_, 0, stream>>>(hp, Mp, rvp, W_rk, W_wk, W_wv, W_er, W_ag,
                                        beta_r, beta_w);
    }
}
